// CoreAttention_9483287790082
// MI455X (gfx1250) — matrix operand order repaired and checked by reading, not compiled, not run on hardware
//
#include <hip/hip_runtime.h>
#include <hip/hip_bf16.h>

typedef __attribute__((ext_vector_type(16))) __bf16 v16bf;
typedef __attribute__((ext_vector_type(8)))  float  v8f;

namespace {
constexpr int BB = 2, SS = 2048, HH = 16, DD = 128;
constexpr float QK_SCALE = 0.08838834764831845f; // 1/sqrt(128)
}

// Flash-attention forward, causal, [B,S,H,D] layout, fp32 in/out, bf16 WMMA math.
// 4 waves / block, each wave owns a 16-row query tile; 32-key blocks staged in LDS.
__global__ __launch_bounds__(128) void fa_fwd_gfx1250(
    const float* __restrict__ Q, const float* __restrict__ K,
    const float* __restrict__ V, float* __restrict__ O)
{
    __shared__ __align__(32) __bf16 sK [32][128];   // [key][d]   8 KB
    __shared__ __align__(32) __bf16 sVt[128][32];   // [d][key]   8 KB (transposed)
    __shared__ __align__(32) __bf16 sP [4][16][32]; // per-wave P relayout scratch

    const int tid    = threadIdx.x;
    const int wave   = tid >> 5;
    const int lane   = tid & 31;
    const int lane16 = lane & 15;
    const int hi     = lane >> 4;          // which 16-lane half

    const int wg_q0 = blockIdx.x * 64;     // first query row of workgroup
    const int qm0   = wg_q0 + wave * 16;   // first query row of this wave's tile

    const size_t sstride = (size_t)HH * DD;                       // stride between seq rows
    const size_t bh_off  = (size_t)blockIdx.z * SS * sstride      // batch
                         + (size_t)blockIdx.y * DD;               // head

    // ---- Q tile -> bf16 A-fragments (16x32 each), pre-scaled by 1/sqrt(D) ----
    // A layout: lane half hi holds K=base..base+15, base = 16*hi.
    v16bf aq[4];
    {
        const float* qrow = Q + bh_off + (size_t)(qm0 + lane16) * sstride;
        const int base = hi * 16;
        #pragma unroll
        for (int c = 0; c < 4; ++c) {
            const int d0 = c * 32;
            #pragma unroll
            for (int i = 0; i < 8; ++i) {
                aq[c][i]     = (__bf16)(qrow[d0 + base + i]      * QK_SCALE);
                aq[c][8 + i] = (__bf16)(qrow[d0 + 8 + base + i] * QK_SCALE);
            }
        }
    }

    // ---- running state: 8 C-tiles (16x16 f32) = 128 output cols; rowmax/rowsum ----
    const v8f vzero = {};
    v8f acc[8];
    #pragma unroll
    for (int nt = 0; nt < 8; ++nt) acc[nt] = vzero;
    float m_i[8], l_i[8];
    #pragma unroll
    for (int r = 0; r < 8; ++r) { m_i[r] = -1e30f; l_i[r] = 0.0f; }

    const int kb_max    = (wg_q0 + 63) >> 5;  // last key block any wave in WG needs
    const int my_kb_max = (qm0  + 15) >> 5;   // last key block this wave needs

    for (int kb = 0; kb <= kb_max; ++kb) {
        __syncthreads();  // LDS reuse guard
        // ---- cooperative stage of K (row-major) and V (transposed) as bf16 ----
        {
            const int k0 = kb * 32;
            #pragma unroll
            for (int j = 0; j < 8; ++j) {
                const int e4  = tid + j * 128;     // 1024 float4 per 32x128 tile
                const int key = e4 >> 5;
                const int c4  = (e4 & 31) * 4;
                const size_t g = bh_off + (size_t)(k0 + key) * sstride + c4;
                const float4 kv = *(const float4*)(K + g);
                sK[key][c4 + 0] = (__bf16)kv.x;
                sK[key][c4 + 1] = (__bf16)kv.y;
                sK[key][c4 + 2] = (__bf16)kv.z;
                sK[key][c4 + 3] = (__bf16)kv.w;
                const float4 vv = *(const float4*)(V + g);
                sVt[c4 + 0][key] = (__bf16)vv.x;
                sVt[c4 + 1][key] = (__bf16)vv.y;
                sVt[c4 + 2][key] = (__bf16)vv.z;
                sVt[c4 + 3][key] = (__bf16)vv.w;
            }
            // prefetch next key block into cache (global_prefetch_b8)
            if (kb < kb_max) {
                const size_t gn = bh_off + (size_t)((kb + 1) * 32 + (tid >> 2)) * sstride
                                + (size_t)(tid & 3) * 32;
                __builtin_prefetch(K + gn, 0, 3);
                __builtin_prefetch(V + gn, 0, 3);
            }
        }
        __syncthreads();

        if (kb > my_kb_max) continue;  // beyond causal range for this wave's tile

        // ---- S = Q K^T : two 16x16 score tiles (keys kb*32..+15 / +16..+31) ----
        // B-fragment: lane holds sK[key = t*16+lane16][d = c*32 + 16*hi .. +15]
        v8f s[2];
        #pragma unroll
        for (int t = 0; t < 2; ++t) {
            v8f c = vzero;
            #pragma unroll
            for (int cc = 0; cc < 4; ++cc) {
                const v16bf bk = *(const v16bf*)&sK[t * 16 + lane16][cc * 32 + hi * 16];
                c = __builtin_amdgcn_wmma_f32_16x16x32_bf16(
                        false, aq[cc], false, bk, (short)0, c, false, false);
            }
            s[t] = c;
        }

        // ---- causal mask + online softmax (C layout: row m=r+8*hi in 16-lane half) ----
        const int key0 = kb * 32;
        #pragma unroll
        for (int r = 0; r < 8; ++r) {
            const int qrow = qm0 + hi * 8 + r;
            if (key0 + lane16      > qrow) s[0][r] = -1e30f;
            if (key0 + 16 + lane16 > qrow) s[1][r] = -1e30f;

            float mx = fmaxf(s[0][r], s[1][r]);
            #pragma unroll
            for (int off = 1; off < 16; off <<= 1)
                mx = fmaxf(mx, __shfl_xor(mx, off, 32));
            const float mnew = fmaxf(m_i[r], mx);

            const float p0 = __expf(s[0][r] - mnew);
            const float p1 = __expf(s[1][r] - mnew);
            s[0][r] = p0; s[1][r] = p1;

            float rs = p0 + p1;
            #pragma unroll
            for (int off = 1; off < 16; off <<= 1)
                rs += __shfl_xor(rs, off, 32);

            const float alpha = __expf(m_i[r] - mnew);
            l_i[r] = alpha * l_i[r] + rs;
            m_i[r] = mnew;
            #pragma unroll
            for (int nt = 0; nt < 8; ++nt) acc[nt][r] *= alpha;
        }

        // ---- P: C layout -> A layout via per-wave LDS scratch ----
        #pragma unroll
        for (int r = 0; r < 8; ++r) {
            sP[wave][hi * 8 + r][lane16]      = (__bf16)s[0][r];
            sP[wave][hi * 8 + r][16 + lane16] = (__bf16)s[1][r];
        }
        v16bf ap;
        {
            const int base = hi * 16;
            #pragma unroll
            for (int i = 0; i < 8; ++i) {
                ap[i]     = sP[wave][lane16][base + i];
                ap[8 + i] = sP[wave][lane16][8 + base + i];
            }
        }

        // ---- O += P V : 8 N-tiles of 16 output columns ----
        // B-fragment: lane holds sVt[d = nt*16+lane16][key = 16*hi .. +15]
        #pragma unroll
        for (int nt = 0; nt < 8; ++nt) {
            const v16bf bv = *(const v16bf*)&sVt[nt * 16 + lane16][hi * 16];
            acc[nt] = __builtin_amdgcn_wmma_f32_16x16x32_bf16(
                          false, ap, false, bv, (short)0, acc[nt], false, false);
        }
    }

    // ---- epilogue: O = acc / l ----
    #pragma unroll
    for (int r = 0; r < 8; ++r) {
        const float inv = 1.0f / l_i[r];
        const int qrow = qm0 + hi * 8 + r;
        float* orow = O + bh_off + (size_t)qrow * sstride;
        #pragma unroll
        for (int nt = 0; nt < 8; ++nt)
            orow[nt * 16 + lane16] = acc[nt][r] * inv;
    }
}

extern "C" void kernel_launch(void* const* d_in, const int* in_sizes, int n_in,
                              void* d_out, int out_size, void* d_ws, size_t ws_size,
                              hipStream_t stream) {
    (void)in_sizes; (void)n_in; (void)out_size; (void)d_ws; (void)ws_size;
    const float* q = (const float*)d_in[0];
    const float* k = (const float*)d_in[1];
    const float* v = (const float*)d_in[2];
    float* o = (float*)d_out;
    dim3 grid(SS / 64, HH, BB);   // (32, 16, 2)
    dim3 block(128);              // 4 wave32
    fa_fwd_gfx1250<<<grid, block, 0, stream>>>(q, k, v, o);
}
